// TransformerBlock_77051713290809
// MI455X (gfx1250) — matrix operand order repaired and checked by reading, not compiled, not run on hardware
//
#include <hip/hip_runtime.h>
#include <hip/hip_bf16.h>

#define BB 2
#define SS 1024
#define PP 79
#define DD 768
#define HH 12
#define DH 64
#define DM 3072
#define NODES (PP + HH + 1)   /* 92 */
#define BS (BB * SS)          /* 2048 */
#define LDA 34                /* padded LDS row stride (halves), 17 dwords */
#define QT 32                 /* q rows per fused-attention block */
#define LDS_S 1034            /* fp32 score row stride: 1034 mod 64 banks -> conflict-free */

typedef __attribute__((ext_vector_type(16))) __bf16 v16bf;
typedef __attribute__((ext_vector_type(2)))  __bf16 v2bf;
typedef __attribute__((ext_vector_type(8)))  float  v8f;

__device__ __forceinline__ unsigned short f2bf(float f) {
  unsigned u = __builtin_bit_cast(unsigned, f);
  u += 0x7FFFu + ((u >> 16) & 1u);          // round-to-nearest-even
  return (unsigned short)(u >> 16);
}

__device__ __forceinline__ v8f zero8() {
  v8f z = {0.f, 0.f, 0.f, 0.f, 0.f, 0.f, 0.f, 0.f};
  return z;
}

// A fragment: 16x32 bf16 from row-major [M][K] source (global or LDS), ld in elements.
// ISA layout: lanes 0-15 / 16-31 split K halves; VGPR j holds a K pair.
__device__ __forceinline__ v16bf load_a_frag(const unsigned short* base, int ld) {
  int lane = threadIdx.x & 31;
  int m = lane & 15, hh = lane >> 4;
  const unsigned short* p = base + m * ld;
  v16bf a;
#pragma unroll
  for (int j = 0; j < 8; ++j) {
    int k = ((j >> 2) << 4) + (hh << 3) + ((j & 3) << 1);
    v2bf t = __builtin_bit_cast(v2bf, *(const unsigned*)(p + k));
    a[2 * j] = t[0];
    a[2 * j + 1] = t[1];
  }
  return a;
}

// Same A layout, but source is fp32 (LDS pattern tile); converts to bf16 on the fly.
__device__ __forceinline__ v16bf load_a_frag_f32(const float* base, int ld) {
  int lane = threadIdx.x & 31;
  int m = lane & 15, hh = lane >> 4;
  const float* p = base + m * ld;
  v16bf a;
#pragma unroll
  for (int j = 0; j < 8; ++j) {
    int k = ((j >> 2) << 4) + (hh << 3) + ((j & 3) << 1);
    a[2 * j] = __builtin_bit_cast(__bf16, f2bf(p[k]));
    a[2 * j + 1] = __builtin_bit_cast(__bf16, f2bf(p[k + 1]));
  }
  return a;
}

// B fragment: 32x16 bf16 from pre-transposed row-major [N][K] source.
__device__ __forceinline__ v16bf load_b_fragT(const unsigned short* base, int ld) {
  int lane = threadIdx.x & 31;
  int n = lane & 15, hh = lane >> 4;
  const unsigned short* p = base + n * ld;
  v16bf b;
#pragma unroll
  for (int j = 0; j < 8; ++j) {
    int k = ((j >> 2) << 4) + (hh << 3) + ((j & 3) << 1);
    v2bf t = __builtin_bit_cast(v2bf, *(const unsigned*)(p + k));
    b[2 * j] = t[0];
    b[2 * j + 1] = t[1];
  }
  return b;
}

__device__ __forceinline__ v8f wmma_bf16(v16bf a, v16bf b, v8f c) {
  return __builtin_amdgcn_wmma_f32_16x16x32_bf16(false, a, false, b, (short)0, c,
                                                 false, false);
}

// ---------------------------------------------------------------- weight prep
__global__ __launch_bounds__(256) void k_conv_qkvo(
    const float* __restrict__ Wq, const float* __restrict__ Wk,
    const float* __restrict__ Wv, const float* __restrict__ Wo,
    unsigned short* __restrict__ WqT, unsigned short* __restrict__ WkT,
    unsigned short* __restrict__ WvT, unsigned short* __restrict__ WoT) {
  int i = blockIdx.x * 256 + threadIdx.x;
  if (i >= HH * DD * DH) return;
  int e = i % DH;
  int d = (i / DH) % DD;
  int h = i / (DH * DD);
  int src = (h * DD + d) * DH + e;          // W_[Q,K,V] : [h][d][e]
  int dstT = (h * DH + e) * DD + d;         // -> [h][e][d]
  WqT[dstT] = f2bf(Wq[src]);
  WkT[dstT] = f2bf(Wk[src]);
  WvT[dstT] = f2bf(Wv[src]);
  WoT[(h * DD + d) * DH + e] = f2bf(Wo[(h * DH + e) * DD + d]);  // [h][e][d] -> [h][d][e]
}

__global__ __launch_bounds__(256) void k_conv_mlp(
    const float* __restrict__ Win, const float* __restrict__ Wout,
    unsigned short* __restrict__ WinT, unsigned short* __restrict__ WoutT) {
  int i = blockIdx.x * 256 + threadIdx.x;
  if (i >= DD * DM) return;
  int j = i % DM;
  int d = i / DM;
  WinT[(size_t)j * DD + d] = f2bf(Win[(size_t)d * DM + j]);    // [d][j] -> [j][d]
  WoutT[(size_t)d * DM + j] = f2bf(Wout[(size_t)j * DD + d]);  // [j][d] -> [d][j]
}

// --------------------------------- fused copy + edge-mask einsum + LN1 stats
__global__ __launch_bounds__(192) void k_mask_copy_stats(
    const float* __restrict__ resid, const float* __restrict__ maskA,
    const float* __restrict__ maskM, float* __restrict__ out,
    float* __restrict__ maskedR, float* __restrict__ mlp_part,
    float* __restrict__ ln1_mean, float* __restrict__ ln1_rstd) {
  __shared__ float smA[PP * HH];
  __shared__ float smM[PP];
  __shared__ float rA[192], rB[192];
  int bs = blockIdx.x;  // flattened b*S+s
  int tid = threadIdx.x;
  for (int idx = tid; idx < PP * HH; idx += 192) smA[idx] = maskA[idx];
  for (int idx = tid; idx < PP; idx += 192) smM[idx] = maskM[idx];
  __syncthreads();
  int d0 = tid * 4;
  float4 acc[HH];
#pragma unroll
  for (int h = 0; h < HH; ++h) acc[h] = make_float4(0.f, 0.f, 0.f, 0.f);
  float4 mp = make_float4(0.f, 0.f, 0.f, 0.f);
  const float* rp = resid + (size_t)bs * PP * DD + d0;
  float* op = out + (size_t)bs * NODES * DD + d0;
  for (int n = 0; n < PP; ++n) {
    float4 r = *(const float4*)(rp + (size_t)n * DD);
    *(float4*)(op + (size_t)n * DD) = r;  // pass-through copy of resid_pre nodes
    float mm = smM[n];
    mp.x += r.x * mm; mp.y += r.y * mm; mp.z += r.z * mm; mp.w += r.w * mm;
#pragma unroll
    for (int h = 0; h < HH; ++h) {
      float wv = smA[n * HH + h];
      acc[h].x += r.x * wv; acc[h].y += r.y * wv;
      acc[h].z += r.z * wv; acc[h].w += r.w * wv;
    }
  }
  *(float4*)(mlp_part + (size_t)bs * DD + d0) = mp;
#pragma unroll
  for (int h = 0; h < HH; ++h)
    *(float4*)(maskedR + ((size_t)bs * HH + h) * DD + d0) = acc[h];
#pragma unroll
  for (int h = 0; h < HH; ++h) {
    float4 a = acc[h];
    rA[tid] = a.x + a.y + a.z + a.w;
    rB[tid] = a.x * a.x + a.y * a.y + a.z * a.z + a.w * a.w;
    __syncthreads();
    for (int off = 128; off > 0; off >>= 1) {
      if (tid < off && tid + off < 192) {
        rA[tid] += rA[tid + off];
        rB[tid] += rB[tid + off];
      }
      __syncthreads();
    }
    if (tid == 0) {
      float mean = rA[0] * (1.f / DD);
      float var = rB[0] * (1.f / DD) - mean * mean;
      ln1_mean[bs * HH + h] = mean;
      ln1_rstd[bs * HH + h] = rsqrtf(var + 1e-5f);
    }
    __syncthreads();
  }
}

// ---------------------- LN1 apply + QKV (WMMA, double-buffered LDS staging)
__global__ __launch_bounds__(256) void k_ln1_qkv(
    const float* __restrict__ maskedR, const float* __restrict__ ln1_mean,
    const float* __restrict__ ln1_rstd, const float* __restrict__ ln1w,
    const float* __restrict__ ln1b, const unsigned short* __restrict__ WqT,
    const unsigned short* __restrict__ WkT, const unsigned short* __restrict__ WvT,
    const float* __restrict__ bQ, const float* __restrict__ bK,
    const float* __restrict__ bV, unsigned short* __restrict__ qb,
    unsigned short* __restrict__ kb, unsigned short* __restrict__ vTb) {
  __shared__ unsigned short lx[2][128 * LDA];
  int h = blockIdx.y;
  int sel = blockIdx.z;  // 0=Q 1=K 2=V
  int g0 = blockIdx.x * 128;
  int tid = threadIdx.x;
  int w = tid >> 5;
  const unsigned short* WT = (sel == 0) ? WqT : (sel == 1) ? WkT : WvT;
  const float* bias = (sel == 0) ? bQ : (sel == 1) ? bK : bV;
  v8f c[4];
#pragma unroll
  for (int i = 0; i < 4; ++i) c[i] = zero8();
  int rl = tid >> 1;           // staged row 0..127
  int kc0 = (tid & 1) * 16;    // 16 contiguous k per thread
  int g = g0 + rl;
  float mu = ln1_mean[g * HH + h];
  float rs = ln1_rstd[g * HH + h];
  const float* xrow = maskedR + ((size_t)g * HH + h) * DD;

  auto stage = [&](int kc, int buf) {
#pragma unroll
    for (int i = 0; i < 16; ++i) {
      int k = kc0 + i;
      float v = (xrow[kc + k] - mu) * rs * ln1w[kc + k] + ln1b[kc + k];
      lx[buf][rl * LDA + k] = f2bf(v);
    }
  };

  stage(0, 0);
  int buf = 0;
  for (int kc = 0; kc < DD; kc += 32, buf ^= 1) {
    __syncthreads();
    if (kc + 32 < DD) stage(kc + 32, buf ^ 1);
    v16bf a = load_a_frag(lx[buf] + (w * 16) * LDA, LDA);
#pragma unroll
    for (int nt = 0; nt < 4; ++nt) {
      v16bf bf = load_b_fragT(WT + ((size_t)(h * DH + nt * 16)) * DD + kc, DD);
      c[nt] = wmma_bf16(a, bf, c[nt]);
    }
  }
  int lane = tid & 31, n = lane & 15, lh = lane >> 4;
  int r0 = g0 + w * 16;
  float scale = (sel == 0) ? 0.125f : 1.f;  // fold 1/sqrt(d_head) into q
#pragma unroll
  for (int nt = 0; nt < 4; ++nt) {
    int e = nt * 16 + n;
    float bv = bias[h * DH + e];
#pragma unroll
    for (int vr = 0; vr < 8; ++vr) {
      int r = r0 + lh * 8 + vr;
      int bI = r >> 10, s = r & (SS - 1);
      unsigned short v = f2bf((c[nt][vr] + bv) * scale);
      if (sel == 2)
        vTb[((size_t)(bI * HH + h) * DH + e) * SS + s] = v;  // v transposed [b][h][e][s]
      else if (sel == 1)
        kb[((size_t)(bI * HH + h) * SS + s) * DH + e] = v;
      else
        qb[((size_t)(bI * HH + h) * SS + s) * DH + e] = v;
    }
  }
}

// ------- fused attention: scores (WMMA) -> LDS, softmax in LDS, z = P@V (WMMA)
__global__ __launch_bounds__(256) void k_attention(
    const unsigned short* __restrict__ qb, const unsigned short* __restrict__ kb,
    const unsigned short* __restrict__ vTb, unsigned short* __restrict__ zb) {
  __shared__ float sS[QT * LDS_S];  // 32 x 1034 fp32 = 132 KB (2 blocks / 320 KB WGP)
  int q0 = blockIdx.x * QT, bh = blockIdx.y;
  int bI = bh / HH, h = bh % HH;
  int tid = threadIdx.x, w = tid >> 5;
  int lane = tid & 31, n = lane & 15, lh = lane >> 4;
  int kend = q0 + QT;  // causal: columns >= kend never contribute

  // ---- phase 1: score tile q0..q0+31 x 0..kend-1 into LDS (scale folded in q)
  v16bf aq[2][2];
#pragma unroll
  for (int qs = 0; qs < 2; ++qs)
#pragma unroll
    for (int ec = 0; ec < 2; ++ec)
      aq[qs][ec] =
          load_a_frag(qb + ((size_t)bh * SS + q0 + qs * 16) * DH + ec * 32, DH);
  for (int kb0 = w * 32; kb0 < kend; kb0 += 256) {
    v8f c[2][2];
#pragma unroll
    for (int qs = 0; qs < 2; ++qs)
#pragma unroll
      for (int nt = 0; nt < 2; ++nt) c[qs][nt] = zero8();
#pragma unroll
    for (int ec = 0; ec < 2; ++ec) {
#pragma unroll
      for (int nt = 0; nt < 2; ++nt) {
        v16bf b =
            load_b_fragT(kb + ((size_t)bh * SS + kb0 + nt * 16) * DH + ec * 32, DH);
#pragma unroll
        for (int qs = 0; qs < 2; ++qs) c[qs][nt] = wmma_bf16(aq[qs][ec], b, c[qs][nt]);
      }
    }
#pragma unroll
    for (int qs = 0; qs < 2; ++qs)
#pragma unroll
      for (int nt = 0; nt < 2; ++nt)
#pragma unroll
        for (int vr = 0; vr < 8; ++vr) {
          int r = qs * 16 + lh * 8 + vr;  // local q row
          int col = kb0 + nt * 16 + n;
          float val = c[qs][nt][vr];
          if (col > q0 + r) val = -1e30f;  // causal mask
          sS[r * LDS_S + col] = val;
        }
  }
  __syncthreads();

  // ---- phase 2: in-place softmax, 8 lanes per row, shfl_xor reductions
  {
    int r = tid >> 3, sub = tid & 7;
    int len = q0 + r + 1;
    float* row = sS + r * LDS_S;
    float mx = -1e30f;
    for (int col = sub; col < len; col += 8) mx = fmaxf(mx, row[col]);
#pragma unroll
    for (int off = 4; off > 0; off >>= 1) mx = fmaxf(mx, __shfl_xor(mx, off, 32));
    float sum = 0.f;
    for (int col = sub; col < len; col += 8) sum += __expf(row[col] - mx);
#pragma unroll
    for (int off = 4; off > 0; off >>= 1) sum += __shfl_xor(sum, off, 32);
    float inv = 1.f / sum;
    for (int col = sub; col < kend; col += 8) {
      float v = row[col];
      row[col] = (col < len) ? __expf(v - mx) * inv : 0.f;
    }
  }
  __syncthreads();

  // ---- phase 3: z = pattern @ v (A from fp32 LDS, cvt to bf16 on load)
  int qs = w & 1, et = w >> 1;  // 8 waves: 2 q-sub x 4 e-tiles
  v8f cz = zero8();
  for (int kc = 0; kc < kend; kc += 32) {
    v16bf a = load_a_frag_f32(sS + (qs * 16) * LDS_S + kc, LDS_S);
    v16bf b = load_b_fragT(vTb + ((size_t)bh * DH + et * 16) * SS + kc, SS);
    cz = wmma_bf16(a, b, cz);
  }
#pragma unroll
  for (int vr = 0; vr < 8; ++vr) {
    int s = q0 + qs * 16 + lh * 8 + vr;
    int e = et * 16 + n;
    zb[(((size_t)bI * SS + s) * HH + h) * DH + e] = f2bf(cz[vr]);
  }
}

// ----------------- per-head W_O projection + fused MLP edge-mask reduction
__global__ __launch_bounds__(256) void k_attn_out(
    const unsigned short* __restrict__ zb, const unsigned short* __restrict__ WoT,
    const float* __restrict__ bO, const float* __restrict__ maskM,
    const float* __restrict__ mlp_part, float* __restrict__ out,
    float* __restrict__ maskedM) {
  int g0 = blockIdx.x * 64, d0 = blockIdx.y * 64;
  int tid = threadIdx.x, w = tid >> 5;
  int rl = (w & 3) * 16, dc = (w >> 2) * 32;
  int lane = tid & 31, n = lane & 15, lh = lane >> 4;
  v8f acc[2] = {zero8(), zero8()};
#pragma unroll 1
  for (int h = 0; h < HH; ++h) {
    v8f t[2] = {zero8(), zero8()};
#pragma unroll
    for (int ec = 0; ec < DH; ec += 32) {
      v16bf a = load_a_frag(zb + ((size_t)(g0 + rl) * HH + h) * DH + ec, HH * DH);
#pragma unroll
      for (int nt = 0; nt < 2; ++nt) {
        v16bf b = load_b_fragT(WoT + ((size_t)h * DD + d0 + dc + nt * 16) * DH + ec, DH);
        t[nt] = wmma_bf16(a, b, t[nt]);
      }
    }
    float mm = maskM[PP + h];
#pragma unroll
    for (int nt = 0; nt < 2; ++nt) {
      int d = d0 + dc + nt * 16 + n;
      float bv = bO[d] * (1.f / HH);
#pragma unroll
      for (int vr = 0; vr < 8; ++vr) {
        int g = g0 + rl + lh * 8 + vr;
        float val = t[nt][vr] + bv;
        out[((size_t)g * NODES + PP + h) * DD + d] = val;  // head node slot
        acc[nt][vr] += mm * val;
      }
    }
  }
#pragma unroll
  for (int nt = 0; nt < 2; ++nt) {
    int d = d0 + dc + nt * 16 + n;
#pragma unroll
    for (int vr = 0; vr < 8; ++vr) {
      int g = g0 + rl + lh * 8 + vr;
      maskedM[(size_t)g * DD + d] = acc[nt][vr] + mlp_part[(size_t)g * DD + d];
    }
  }
}

// ------------------------------------------------------------- LN2 stats
__global__ __launch_bounds__(256) void k_ln2_stats(const float* __restrict__ maskedM,
                                                   float* __restrict__ mean,
                                                   float* __restrict__ rstd) {
  __shared__ float rA[256], rB[256];
  int g = blockIdx.x, tid = threadIdx.x;
  const float* xp = maskedM + (size_t)g * DD;
  float s1 = 0.f, s2 = 0.f;
  for (int k = tid; k < DD; k += 256) {
    float v = xp[k];
    s1 += v;
    s2 += v * v;
  }
  rA[tid] = s1;
  rB[tid] = s2;
  __syncthreads();
  for (int off = 128; off > 0; off >>= 1) {
    if (tid < off) {
      rA[tid] += rA[tid + off];
      rB[tid] += rB[tid + off];
    }
    __syncthreads();
  }
  if (tid == 0) {
    float mu = rA[0] * (1.f / DD);
    float var = rB[0] * (1.f / DD) - mu * mu;
    mean[g] = mu;
    rstd[g] = rsqrtf(var + 1e-5f);
  }
}

// ------------- MLP in: LN2 + GEMM + GELU (WMMA, double-buffered staging)
__global__ __launch_bounds__(256) void k_mlp_in(
    const float* __restrict__ maskedM, const float* __restrict__ mean,
    const float* __restrict__ rstd, const float* __restrict__ ln2w,
    const float* __restrict__ ln2b, const unsigned short* __restrict__ WinT,
    const float* __restrict__ bin, unsigned short* __restrict__ hidden) {
  __shared__ unsigned short lx[2][64 * LDA];
  int g0 = blockIdx.x * 64, j0 = blockIdx.y * 64;
  int tid = threadIdx.x, w = tid >> 5;
  int rl = tid >> 2;          // staged row 0..63
  int kk0 = (tid & 3) * 8;    // 8 contiguous k per thread
  int g = g0 + rl;
  float mu = mean[g], rs = rstd[g];
  int wr = (w & 3) * 16, wc = (w >> 2) * 32;
  v8f c[2] = {zero8(), zero8()};

  auto stage = [&](int kc, int buf) {
#pragma unroll
    for (int i = 0; i < 8; ++i) {
      int k = kk0 + i;
      float v = (maskedM[(size_t)g * DD + kc + k] - mu) * rs * ln2w[kc + k] + ln2b[kc + k];
      lx[buf][rl * LDA + k] = f2bf(v);
    }
  };

  stage(0, 0);
  int buf = 0;
  for (int kc = 0; kc < DD; kc += 32, buf ^= 1) {
    __syncthreads();
    if (kc + 32 < DD) stage(kc + 32, buf ^ 1);
    v16bf a = load_a_frag(lx[buf] + wr * LDA, LDA);
#pragma unroll
    for (int nt = 0; nt < 2; ++nt) {
      v16bf b = load_b_fragT(WinT + ((size_t)(j0 + wc + nt * 16)) * DD + kc, DD);
      c[nt] = wmma_bf16(a, b, c[nt]);
    }
  }
  int lane = tid & 31, n = lane & 15, lh = lane >> 4;
#pragma unroll
  for (int nt = 0; nt < 2; ++nt) {
    int j = j0 + wc + nt * 16 + n;
    float bj = bin[j];
#pragma unroll
    for (int vr = 0; vr < 8; ++vr) {
      int gg = g0 + wr + lh * 8 + vr;
      float x = c[nt][vr] + bj;
      float inner = 0.7978845608028654f * (x + 0.044715f * x * x * x);
      float y = 0.5f * x * (1.f + tanhf(inner));
      hidden[(size_t)gg * DM + j] = f2bf(y);
    }
  }
}

// ------------------------------------------------- MLP out GEMM (WMMA)
__global__ __launch_bounds__(256) void k_mlp_out(
    const unsigned short* __restrict__ hidden, const unsigned short* __restrict__ WoutT,
    const float* __restrict__ bout, float* __restrict__ out) {
  int g0 = blockIdx.x * 64, d0 = blockIdx.y * 64;
  int tid = threadIdx.x, w = tid >> 5;
  int wr = (w & 3) * 16, wc = (w >> 2) * 32;
  v8f c[2] = {zero8(), zero8()};
  for (int kc = 0; kc < DM; kc += 32) {
    if (kc + 32 < DM) {  // prefetch next K chunk (global_prefetch_b8)
      __builtin_prefetch(hidden + (size_t)(g0 + wr) * DM + kc + 32, 0, 1);
      __builtin_prefetch(WoutT + (size_t)(d0 + wc) * DM + kc + 32, 0, 1);
    }
    v16bf a = load_a_frag(hidden + (size_t)(g0 + wr) * DM + kc, DM);
#pragma unroll
    for (int nt = 0; nt < 2; ++nt) {
      v16bf b = load_b_fragT(WoutT + (size_t)(d0 + wc + nt * 16) * DM + kc, DM);
      c[nt] = wmma_bf16(a, b, c[nt]);
    }
  }
  int lane = tid & 31, n = lane & 15, lh = lane >> 4;
#pragma unroll
  for (int nt = 0; nt < 2; ++nt) {
    int d = d0 + wc + nt * 16 + n;
    float bv = bout[d];
#pragma unroll
    for (int vr = 0; vr < 8; ++vr) {
      int g = g0 + wr + lh * 8 + vr;
      out[((size_t)g * NODES + PP + HH) * DD + d] = c[nt][vr] + bv;  // node 91
    }
  }
}

extern "C" void kernel_launch(void* const* d_in, const int* in_sizes, int n_in,
                              void* d_out, int out_size, void* d_ws, size_t ws_size,
                              hipStream_t stream) {
  (void)in_sizes; (void)n_in; (void)out_size; (void)ws_size;
  const float* resid = (const float*)d_in[0];
  const float* maskA = (const float*)d_in[1];
  const float* maskM = (const float*)d_in[2];
  const float* ln1w = (const float*)d_in[3];
  const float* ln1b = (const float*)d_in[4];
  const float* ln2w = (const float*)d_in[5];
  const float* ln2b = (const float*)d_in[6];
  const float* Wq = (const float*)d_in[7];
  const float* bQ = (const float*)d_in[8];
  const float* Wk = (const float*)d_in[9];
  const float* bK = (const float*)d_in[10];
  const float* Wv = (const float*)d_in[11];
  const float* bV = (const float*)d_in[12];
  const float* Wo = (const float*)d_in[13];
  const float* bO = (const float*)d_in[14];
  const float* Win = (const float*)d_in[15];
  const float* bin = (const float*)d_in[16];
  const float* Wout = (const float*)d_in[17];
  const float* bout = (const float*)d_in[18];
  float* out = (float*)d_out;

  char* wp = (char*)d_ws;
  auto alloc = [&](size_t bytes) -> void* {
    void* p = (void*)wp;
    wp += (bytes + 255) & ~(size_t)255;
    return p;
  };
  float* maskedR = (float*)alloc((size_t)BS * HH * DD * 4);
  float* ln1_mean = (float*)alloc((size_t)BS * HH * 4);
  float* ln1_rstd = (float*)alloc((size_t)BS * HH * 4);
  float* mlp_part = (float*)alloc((size_t)BS * DD * 4);
  unsigned short* qb = (unsigned short*)alloc((size_t)BB * HH * SS * DH * 2);
  unsigned short* kb = (unsigned short*)alloc((size_t)BB * HH * SS * DH * 2);
  unsigned short* vTb = (unsigned short*)alloc((size_t)BB * HH * DH * SS * 2);
  unsigned short* zb = (unsigned short*)alloc((size_t)BS * HH * DH * 2);
  float* maskedM = (float*)alloc((size_t)BS * DD * 4);
  float* ln2_mean = (float*)alloc((size_t)BS * 4);
  float* ln2_rstd = (float*)alloc((size_t)BS * 4);
  unsigned short* hidden = (unsigned short*)alloc((size_t)BS * DM * 2);
  unsigned short* WqT = (unsigned short*)alloc((size_t)HH * DH * DD * 2);
  unsigned short* WkT = (unsigned short*)alloc((size_t)HH * DH * DD * 2);
  unsigned short* WvT = (unsigned short*)alloc((size_t)HH * DH * DD * 2);
  unsigned short* WoT = (unsigned short*)alloc((size_t)HH * DD * DH * 2);
  unsigned short* WinT = (unsigned short*)alloc((size_t)DM * DD * 2);
  unsigned short* WoutT = (unsigned short*)alloc((size_t)DD * DM * 2);

  k_conv_qkvo<<<(HH * DD * DH + 255) / 256, 256, 0, stream>>>(Wq, Wk, Wv, Wo, WqT, WkT,
                                                              WvT, WoT);
  k_conv_mlp<<<(DD * DM + 255) / 256, 256, 0, stream>>>(Win, Wout, WinT, WoutT);
  k_mask_copy_stats<<<BS, 192, 0, stream>>>(resid, maskA, maskM, out, maskedR, mlp_part,
                                            ln1_mean, ln1_rstd);
  k_ln1_qkv<<<dim3(BS / 128, HH, 3), 256, 0, stream>>>(maskedR, ln1_mean, ln1_rstd, ln1w,
                                                       ln1b, WqT, WkT, WvT, bQ, bK, bV,
                                                       qb, kb, vTb);
  k_attention<<<dim3(SS / QT, BB * HH), 256, 0, stream>>>(qb, kb, vTb, zb);
  k_attn_out<<<dim3(BS / 64, DD / 64), 256, 0, stream>>>(zb, WoT, bO, maskM, mlp_part,
                                                         out, maskedM);
  k_ln2_stats<<<BS, 256, 0, stream>>>(maskedM, ln2_mean, ln2_rstd);
  k_mlp_in<<<dim3(BS / 64, DM / 64), 256, 0, stream>>>(maskedM, ln2_mean, ln2_rstd, ln2w,
                                                       ln2b, WinT, bin, hidden);
  k_mlp_out<<<dim3(BS / 64, DD / 64), 256, 0, stream>>>(hidden, WoutT, bout, out);
}
